// GraphSwinEncoder_9148280341100
// MI455X (gfx1250) — matrix operand order repaired and checked by reading, not compiled, not run on hardware
//
#include <hip/hip_runtime.h>
#include <hip/hip_bf16.h>

// ---------------------------------------------------------------------------
// GATv2 x2 on 2048 independent 7x7-grid graphs, bf16 WMMA GEMMs (f32 accum).
//
// ws layout (bf16 elements):
//   [0]                      Wl1 packed  (768x512)  B-fragment layout
//   [768*512]                Wr1 packed  (768x512)
//   [2*768*512]              Wl2 packed  (512x256)
//   [2*768*512 + 512*256]    Wr2 packed  (512x256)
//   [2*768*512 + 2*512*256]  h intermediate, 2048 * 49 * 512 bf16 (~103 MB)
// ---------------------------------------------------------------------------

typedef __attribute__((ext_vector_type(16))) __bf16 v16bf;
typedef __attribute__((ext_vector_type(8)))  __bf16 v8bf;
typedef __attribute__((ext_vector_type(8)))  float  v8f;

#define NGRAPH 2048
#define NNODE  49
#define FIN    768
#define H1     4
#define C1     128
#define HC1    512
#define C2     256
#define KT1    24      // 768/32 k-tiles, layer 1
#define KT2    16      // 512/32 k-tiles, layer 2

#define WS_WL1P 0
#define WS_WR1P (768 * 512)
#define WS_WL2P (2 * 768 * 512)
#define WS_WR2P (2 * 768 * 512 + 512 * 256)
#define WS_HBUF (2 * 768 * 512 + 2 * 512 * 256)

__device__ __forceinline__ v8f wmma_bf16(v16bf a, v16bf b, v8f c) {
  return __builtin_amdgcn_wmma_f32_16x16x32_bf16(false, a, false, b, (short)0, c,
                                                 false, false);
}

// Build a 16x32 bf16 A fragment from fp32 row data.
// lane<16: K {0..7,16..23}; lane>=16: K {8..15,24..31} (khalf folded into ap).
__device__ __forceinline__ v16bf build_a_f32(const float* __restrict__ ap) {
  const float4* p = reinterpret_cast<const float4*>(ap);
  float4 f0 = p[0], f1 = p[1], f2 = p[4], f3 = p[5];
  v16bf a;
  a[0]  = (__bf16)f0.x; a[1]  = (__bf16)f0.y; a[2]  = (__bf16)f0.z; a[3]  = (__bf16)f0.w;
  a[4]  = (__bf16)f1.x; a[5]  = (__bf16)f1.y; a[6]  = (__bf16)f1.z; a[7]  = (__bf16)f1.w;
  a[8]  = (__bf16)f2.x; a[9]  = (__bf16)f2.y; a[10] = (__bf16)f2.z; a[11] = (__bf16)f2.w;
  a[12] = (__bf16)f3.x; a[13] = (__bf16)f3.y; a[14] = (__bf16)f3.z; a[15] = (__bf16)f3.w;
  return a;
}

// Build an A fragment from bf16 row data (layer-2 input h).
__device__ __forceinline__ v16bf build_a_bf16(const __bf16* __restrict__ ap) {
  v8bf lo = *reinterpret_cast<const v8bf*>(ap);
  v8bf hi = *reinterpret_cast<const v8bf*>(ap + 16);
  v16bf a;
#pragma unroll
  for (int t = 0; t < 8; ++t) { a[t] = lo[t]; a[8 + t] = hi[t]; }
  return a;
}

// ---------------------------------------------------------------------------
// Weight repack: fp32 row-major [K, N] -> bf16 WMMA B-fragment layout.
// idx = (((ntile*Ktiles + ktile)*32 + lane)*16 + e)
//   lane 0..15  -> col = ntile*16 + lane,     K = ktile*32 + (e&7) + (e>>3)*16
//   lane 16..31 -> col = ntile*16 + lane-16,  K = ktile*32 + 8 + (e&7) + (e>>3)*16
// ---------------------------------------------------------------------------
__device__ inline void pack_one(const float* __restrict__ src, __bf16* __restrict__ dst,
                                int Ktiles, int Nw, int idx) {
  int e    = idx & 15;
  int lane = (idx >> 4) & 31;
  int t    = idx >> 9;
  int ktile = t % Ktiles;
  int ntile = t / Ktiles;
  int col = ntile * 16 + (lane & 15);
  int k   = ktile * 32 + ((lane >> 4) << 3) + (e & 7) + ((e >> 3) << 4);
  dst[idx] = (__bf16)src[k * Nw + col];
}

__global__ void pack_weights_kernel(const float* __restrict__ Wl1, const float* __restrict__ Wr1,
                                    const float* __restrict__ Wl2, const float* __restrict__ Wr2,
                                    __bf16* __restrict__ ws) {
  const int S1 = 768 * 512;
  const int S2 = 512 * 256;
  int idx = blockIdx.x * blockDim.x + threadIdx.x;
  if (idx < S1)                   pack_one(Wl1, ws + WS_WL1P, 24, 512, idx);
  else if (idx < 2 * S1)          pack_one(Wr1, ws + WS_WR1P, 24, 512, idx - S1);
  else if (idx < 2 * S1 + S2)     pack_one(Wl2, ws + WS_WL2P, 16, 256, idx - 2 * S1);
  else if (idx < 2 * S1 + 2 * S2) pack_one(Wr2, ws + WS_WR2P, 16, 256, idx - (2 * S1 + S2));
}

// Load 4 B fragments at compile-time-constant strides (immediate offsets).
#define LOAD_B4(dst, basep, KSTRIDE)                                          \
  do {                                                                        \
    (dst)[0] = *reinterpret_cast<const v16bf*>((basep));                      \
    (dst)[1] = *reinterpret_cast<const v16bf*>((basep) + (KSTRIDE) * 512);    \
    (dst)[2] = *reinterpret_cast<const v16bf*>((basep) + 2 * (KSTRIDE) * 512);\
    (dst)[3] = *reinterpret_cast<const v16bf*>((basep) + 3 * (KSTRIDE) * 512);\
  } while (0)

#define WMMA8(a0_, a1_, bb)                                                   \
  do {                                                                        \
    _Pragma("unroll")                                                         \
    for (int j = 0; j < 4; ++j) {                                             \
      acc0[j] = wmma_bf16((a0_), (bb)[j], acc0[j]);                           \
      acc1[j] = wmma_bf16((a1_), (bb)[j], acc1[j]);                           \
    }                                                                         \
  } while (0)

// ---------------------------------------------------------------------------
// Layer 1.  8 waves: wave = (nquad<<2)|(sel<<1)|mhalf.
// Each wave: 2 m-tiles (mhalf) x 4 n-tiles (nquad within head slice),
// 8 accumulators, double-buffered B, pipelined over 24 k-tiles.
// ---------------------------------------------------------------------------
__global__ __launch_bounds__(256) void gat_layer1_kernel(
    const float* __restrict__ x, const __bf16* __restrict__ ws,
    const float* __restrict__ att1, const float* __restrict__ b1,
    __bf16* __restrict__ hbuf) {
  __shared__ float sXL[NNODE * C1];
  __shared__ float sXR[NNODE * C1];
  __shared__ float sAl[NNODE * 5];
  __shared__ int   sNb[NNODE * 5];
  __shared__ int   sNn[NNODE];

  const int g = blockIdx.x;
  const float* xg = x + (size_t)g * NNODE * FIN;
  __bf16* hg = hbuf + (size_t)g * NNODE * HC1;

  const int tid   = threadIdx.x;
  const int wave  = tid >> 5, lane = tid & 31;
  const int mhalf = wave & 1;
  const int sel   = (wave >> 1) & 1;
  const int nquad = wave >> 2;            // 0..1
  const __bf16* Wp = ws + (sel ? WS_WR1P : WS_WL1P);
  float* sD = sel ? sXR : sXL;

  const int khalf = lane >> 4;
  const int colD  = lane & 15;
  const int row0  = mhalf * 32 + (lane & 15);          // always < 48: valid
  const int row1r = row0 + 16;
  const int row1  = row1r < NNODE ? row1r : 0;         // clamp; D rows >=49 discarded
  const float* ap0 = xg + row0 * FIN + khalf * 8;
  const float* ap1 = xg + row1 * FIN + khalf * 8;

  for (int head = 0; head < H1; ++head) {
    const int n0 = head * 8 + nquad * 4;               // n-tile base in this weight
    v8f acc0[4], acc1[4];
#pragma unroll
    for (int j = 0; j < 4; ++j)
#pragma unroll
      for (int r = 0; r < 8; ++r) { acc0[j][r] = 0.0f; acc1[j][r] = 0.0f; }

    const __bf16* bq = Wp + (size_t)n0 * KT1 * 512 + (size_t)lane * 16;
    v16bf bA[4], bB[4];
    LOAD_B4(bA, bq, KT1);                              // k-tile 0
    for (int k = 0; k < KT1 - 2; k += 2) {
      v16bf a0 = build_a_f32(ap0 + k * 32);
      v16bf a1 = build_a_f32(ap1 + k * 32);
      LOAD_B4(bB, bq + (size_t)(k + 1) * 512, KT1);
      WMMA8(a0, a1, bA);
      a0 = build_a_f32(ap0 + (k + 1) * 32);
      a1 = build_a_f32(ap1 + (k + 1) * 32);
      LOAD_B4(bA, bq + (size_t)(k + 2) * 512, KT1);
      WMMA8(a0, a1, bB);
    }
    {   // epilogue: k-tiles KT1-2, KT1-1
      const int k = KT1 - 2;
      v16bf a0 = build_a_f32(ap0 + k * 32);
      v16bf a1 = build_a_f32(ap1 + k * 32);
      LOAD_B4(bB, bq + (size_t)(k + 1) * 512, KT1);
      WMMA8(a0, a1, bA);
      a0 = build_a_f32(ap0 + (k + 1) * 32);
      a1 = build_a_f32(ap1 + (k + 1) * 32);
      WMMA8(a0, a1, bB);
    }

    // store D tiles (fp32) into head-local LDS [49 x 128]
#pragma unroll
    for (int j = 0; j < 4; ++j) {
      const int col = (nquad * 4 + j) * 16 + colD;
      const int rb0 = (mhalf * 2) * 16 + khalf * 8;
#pragma unroll
      for (int r = 0; r < 8; ++r) {
        const int row = rb0 + r;
        if (row < NNODE) sD[row * C1 + col] = acc0[j][r];
      }
      const int rb1 = (mhalf * 2 + 1) * 16 + khalf * 8;
#pragma unroll
      for (int r = 0; r < 8; ++r) {
        const int row = rb1 + r;
        if (row < NNODE) sD[row * C1 + col] = acc1[j][r];
      }
    }
    __syncthreads();

    // ---- attention scores + softmax (one thread per target node) ----
    if (tid < NNODE) {
      const int i = tid, rr = i / 7, cc = i % 7;
      int nb[5]; int nn = 0;
      nb[nn++] = i;                       // self-loop
      if (rr > 0) nb[nn++] = i - 7;
      if (rr < 6) nb[nn++] = i + 7;
      if (cc > 0) nb[nn++] = i - 1;
      if (cc < 6) nb[nn++] = i + 1;
      float sc[5] = {0.0f, 0.0f, 0.0f, 0.0f, 0.0f};
      for (int c2 = 0; c2 < C1; ++c2) {
        const float av = att1[head * C1 + c2];
        const float xr = sXR[i * C1 + c2];
        for (int e = 0; e < nn; ++e) {
          float v = sXL[nb[e] * C1 + c2] + xr;
          v = v > 0.0f ? v : 0.2f * v;    // LeakyReLU(0.2)
          sc[e] += av * v;
        }
      }
      float m = sc[0];
      for (int e = 1; e < nn; ++e) m = fmaxf(m, sc[e]);
      float den = 0.0f;
      for (int e = 0; e < nn; ++e) { sc[e] = expf(sc[e] - m); den += sc[e]; }
      const float inv = 1.0f / den;
      for (int e = 0; e < nn; ++e) { sAl[i * 5 + e] = sc[e] * inv; sNb[i * 5 + e] = nb[e]; }
      sNn[i] = nn;
    }
    __syncthreads();

    // ---- aggregate + bias + ELU -> bf16 h (all 256 threads) ----
    for (int idx = tid; idx < NNODE * C1; idx += 256) {
      const int i = idx >> 7, c2 = idx & 127;
      const int nn = sNn[i];
      float o = 0.0f;
      for (int e = 0; e < nn; ++e) o += sAl[i * 5 + e] * sXL[sNb[i * 5 + e] * C1 + c2];
      o += b1[head * C1 + c2];
      o = o > 0.0f ? o : (expf(o) - 1.0f);  // ELU
      hg[i * HC1 + head * C1 + c2] = (__bf16)o;
    }
    __syncthreads();
  }
}

// ---------------------------------------------------------------------------
// Layer 2: yl = h@Wl2, yr = h@Wr2 (bf16 WMMA), single-head attention over
// 256 channels, deterministic per-channel mean pool + bias.
// ---------------------------------------------------------------------------
__global__ __launch_bounds__(256) void gat_layer2_kernel(
    const __bf16* __restrict__ ws, const float* __restrict__ att2,
    const float* __restrict__ b2, float* __restrict__ out) {
  __shared__ __bf16 sYL[NNODE * C2];
  __shared__ __bf16 sYR[NNODE * C2];
  __shared__ float  sAl[NNODE * 5];
  __shared__ int    sNb[NNODE * 5];
  __shared__ int    sNn[NNODE];

  const int g = blockIdx.x;
  const __bf16* hg = ws + WS_HBUF + (size_t)g * NNODE * HC1;

  const int tid   = threadIdx.x;
  const int wave  = tid >> 5, lane = tid & 31;
  const int mhalf = wave & 1;
  const int sel   = (wave >> 1) & 1;
  const int nquad = wave >> 2;            // 0..1
  const __bf16* Wp = ws + (sel ? WS_WR2P : WS_WL2P);
  __bf16* sD = sel ? sYR : sYL;

  const int khalf = lane >> 4;
  const int colD  = lane & 15;
  const int row0  = mhalf * 32 + (lane & 15);
  const int row1r = row0 + 16;
  const int row1  = row1r < NNODE ? row1r : 0;
  const __bf16* ap0 = hg + row0 * HC1 + khalf * 8;
  const __bf16* ap1 = hg + row1 * HC1 + khalf * 8;

  for (int grp = 0; grp < 2; ++grp) {
    const int n0 = nquad * 8 + grp * 4;
    v8f acc0[4], acc1[4];
#pragma unroll
    for (int j = 0; j < 4; ++j)
#pragma unroll
      for (int r = 0; r < 8; ++r) { acc0[j][r] = 0.0f; acc1[j][r] = 0.0f; }

    const __bf16* bq = Wp + (size_t)n0 * KT2 * 512 + (size_t)lane * 16;
    v16bf bA[4], bB[4];
    LOAD_B4(bA, bq, KT2);
    for (int k = 0; k < KT2 - 2; k += 2) {
      v16bf a0 = build_a_bf16(ap0 + k * 32);
      v16bf a1 = build_a_bf16(ap1 + k * 32);
      LOAD_B4(bB, bq + (size_t)(k + 1) * 512, KT2);
      WMMA8(a0, a1, bA);
      a0 = build_a_bf16(ap0 + (k + 1) * 32);
      a1 = build_a_bf16(ap1 + (k + 1) * 32);
      LOAD_B4(bA, bq + (size_t)(k + 2) * 512, KT2);
      WMMA8(a0, a1, bB);
    }
    {
      const int k = KT2 - 2;
      v16bf a0 = build_a_bf16(ap0 + k * 32);
      v16bf a1 = build_a_bf16(ap1 + k * 32);
      LOAD_B4(bB, bq + (size_t)(k + 1) * 512, KT2);
      WMMA8(a0, a1, bA);
      a0 = build_a_bf16(ap0 + (k + 1) * 32);
      a1 = build_a_bf16(ap1 + (k + 1) * 32);
      WMMA8(a0, a1, bB);
    }

#pragma unroll
    for (int j = 0; j < 4; ++j) {
      const int col = (n0 + j) * 16 + colD;
      const int rb0 = (mhalf * 2) * 16 + khalf * 8;
#pragma unroll
      for (int r = 0; r < 8; ++r) {
        const int row = rb0 + r;
        if (row < NNODE) sD[row * C2 + col] = (__bf16)acc0[j][r];
      }
      const int rb1 = (mhalf * 2 + 1) * 16 + khalf * 8;
#pragma unroll
      for (int r = 0; r < 8; ++r) {
        const int row = rb1 + r;
        if (row < NNODE) sD[row * C2 + col] = (__bf16)acc1[j][r];
      }
    }
  }
  __syncthreads();

  if (tid < NNODE) {
    const int i = tid, rr = i / 7, cc = i % 7;
    int nb[5]; int nn = 0;
    nb[nn++] = i;
    if (rr > 0) nb[nn++] = i - 7;
    if (rr < 6) nb[nn++] = i + 7;
    if (cc > 0) nb[nn++] = i - 1;
    if (cc < 6) nb[nn++] = i + 1;
    float sc[5] = {0.0f, 0.0f, 0.0f, 0.0f, 0.0f};
    for (int c2 = 0; c2 < C2; ++c2) {
      const float av = att2[c2];
      const float xr = (float)sYR[i * C2 + c2];
      for (int e = 0; e < nn; ++e) {
        float v = (float)sYL[nb[e] * C2 + c2] + xr;
        v = v > 0.0f ? v : 0.2f * v;
        sc[e] += av * v;
      }
    }
    float m = sc[0];
    for (int e = 1; e < nn; ++e) m = fmaxf(m, sc[e]);
    float den = 0.0f;
    for (int e = 0; e < nn; ++e) { sc[e] = expf(sc[e] - m); den += sc[e]; }
    const float inv = 1.0f / den;
    for (int e = 0; e < nn; ++e) { sAl[i * 5 + e] = sc[e] * inv; sNb[i * 5 + e] = nb[e]; }
    sNn[i] = nn;
  }
  __syncthreads();

  // per-channel aggregate + mean pool (deterministic; 256 threads = 256 ch)
  {
    const int c2 = tid;
    float s = 0.0f;
    for (int i = 0; i < NNODE; ++i) {
      const int nn = sNn[i];
      for (int e = 0; e < nn; ++e)
        s += sAl[i * 5 + e] * (float)sYL[sNb[i * 5 + e] * C2 + c2];
    }
    out[(size_t)g * C2 + c2] = s * (1.0f / 49.0f) + b2[c2];
  }
}

// ---------------------------------------------------------------------------
extern "C" void kernel_launch(void* const* d_in, const int* in_sizes, int n_in,
                              void* d_out, int out_size, void* d_ws, size_t ws_size,
                              hipStream_t stream) {
  (void)in_sizes; (void)n_in; (void)out_size; (void)ws_size;
  const float* x    = (const float*)d_in[0];
  const float* Wl1  = (const float*)d_in[1];
  const float* Wr1  = (const float*)d_in[2];
  const float* att1 = (const float*)d_in[3];
  const float* b1   = (const float*)d_in[4];
  const float* Wl2  = (const float*)d_in[5];
  const float* Wr2  = (const float*)d_in[6];
  const float* att2 = (const float*)d_in[7];
  const float* b2   = (const float*)d_in[8];
  __bf16* ws = (__bf16*)d_ws;
  float* out = (float*)d_out;

  pack_weights_kernel<<<4096, 256, 0, stream>>>(Wl1, Wr1, Wl2, Wr2, ws);
  gat_layer1_kernel<<<NGRAPH, 256, 0, stream>>>(x, ws, att1, b1, ws + WS_HBUF);
  gat_layer2_kernel<<<NGRAPH, 256, 0, stream>>>(ws, att2, b2, out);
}
